// ContinuousFilterConv_63342177681981
// MI455X (gfx1250) — matrix operand order repaired and checked by reading, not compiled, not run on hardware
//
#include <hip/hip_runtime.h>
#include <hip/hip_bf16.h>
#include <stdint.h>

// CDNA5 / gfx1250 continuous-filter convolution.
// bf16 WMMA (f32 acc) filter MLP, fused gather+reduce.
// Round 4: fix async builtin pointer types (v4i AS1* / v4i AS3*).

typedef __attribute__((ext_vector_type(16))) __bf16 v16bf;
typedef __attribute__((ext_vector_type(8)))  float  v8f;
typedef __attribute__((ext_vector_type(4)))  int    v4i;

#define BDIM    256
#define B_FRM   32
#define A_ATOMS 512
#define NNBR    64
#define G_DIM   64
#define F_DIM   128
#define ITER    4                     // (b,a) pairs per block

#define W1T_ELEMS (F_DIM * G_DIM)     // 8192  bf16  (K-contiguous, [f][g])
#define W2T_ELEMS (F_DIM * F_DIM)     // 16384 bf16  (K-contiguous, [f][c])
#define WS_BYTES  ((W1T_ELEMS + W2T_ELEMS) * 2)   // 48 KB

#define AS1 __attribute__((address_space(1)))
#define AS3 __attribute__((address_space(3)))

#if defined(__has_builtin)
#  if __has_builtin(__builtin_amdgcn_tanhf)
#    define TANHF(x) __builtin_amdgcn_tanhf(x)
#  else
#    define TANHF(x) tanhf(x)
#  endif
#  if __has_builtin(__builtin_amdgcn_global_load_async_to_lds_b128) && \
      __has_builtin(__builtin_amdgcn_s_wait_asynccnt)
#    define HAVE_ASYNC_LDS 1
#  else
#    define HAVE_ASYNC_LDS 0
#  endif
#else
#  define TANHF(x) tanhf(x)
#  define HAVE_ASYNC_LDS 0
#endif

union Frag { v16bf v; uint32_t u[8]; };

// A-matrix 16x32 bf16 fragment from row-major LDS (ISA 7.12.2):
// lanes 0-15: M=lane, dword j holds K = {0,2,4,6,16,18,20,22}[j]..+1
// lanes 16-31: same rows, K shifted by +8.  -> two ds_load_b128 per lane.
__device__ __forceinline__ v16bf load_a_frag(const __bf16* base, int row0, int k0,
                                             int kstride, int lane) {
  Frag fr;
  const int m = lane & 15, half = lane >> 4;
  const __bf16* p = base + (size_t)(row0 + m) * kstride + k0 + half * 8;
#pragma unroll
  for (int j = 0; j < 8; ++j) {
    const int kb = (j < 4) ? (2 * j) : (8 + 2 * j);   // 0,2,4,6,16,18,20,22
    fr.u[j] = *(const uint32_t*)(p + kb);
  }
  return fr.v;
}

// B-matrix 32x16 bf16 fragment from K-contiguous (transposed) LDS:
// lanes 0-15: N=lane, dword j holds K = {0,2,4,6,16,18,20,22}[j]..+1 ; lanes 16-31: K += 8.
__device__ __forceinline__ v16bf load_b_frag(const __bf16* baseT, int n0, int k0,
                                             int kstride, int lane) {
  Frag fr;
  const int n = lane & 15, half = lane >> 4;
  const __bf16* p = baseT + (size_t)(n0 + n) * kstride + k0 + half * 8;
#pragma unroll
  for (int j = 0; j < 8; ++j) fr.u[j] = *(const uint32_t*)(p + ((j < 4) ? (2 * j) : (8 + 2 * j)));
  return fr.v;
}

// ---- One-shot weight conversion: W1[G][F] -> ws[f*64+g] bf16,
//                                  W2[C][F] -> ws[8192 + f*128+c] bf16.
extern "C" __global__ __launch_bounds__(BDIM)
void cfconv_prep_weights(const float* __restrict__ W1,
                         const float* __restrict__ W2,
                         __bf16* __restrict__ ws) {
  const int tid = threadIdx.x + blockIdx.x * BDIM;
  const int nth = gridDim.x * BDIM;
  for (int i = tid; i < W1T_ELEMS; i += nth) {
    const int g = i >> 7, f = i & 127;
    ws[f * G_DIM + g] = (__bf16)W1[i];
  }
  for (int i = tid; i < W2T_ELEMS; i += nth) {
    const int c = i >> 7, f = i & 127;
    ws[W1T_ELEMS + f * F_DIM + c] = (__bf16)W2[i];
  }
}

template <bool PREP>
__global__ __launch_bounds__(BDIM)
void cfconv_wmma_bf16(const float* __restrict__ features,
                      const float* __restrict__ rbf,
                      const long long* __restrict__ nlist,
                      const float* __restrict__ W1, const float* __restrict__ b1,
                      const float* __restrict__ W2, const float* __restrict__ b2,
                      const __bf16* __restrict__ wsw,
                      float* __restrict__ out) {
  extern __shared__ char smem_raw[];
  __bf16* W1t = (__bf16*)smem_raw;            // [F=128][G=64]  K-contiguous
  __bf16* W2t = W1t + W1T_ELEMS;              // [F=128][K=128] K-contiguous
  __bf16* Ab  = W2t + W2T_ELEMS;              // [N=64][G=64]   rbf tile, bf16
  __bf16* H1  = Ab  + NNBR * G_DIM;           // [N=64][F=128]  tanh layer out
  float*  b1s = (float*)(H1 + NNBR * F_DIM);  // [128]
  float*  b2s = b1s + F_DIM;                  // [128]
  float*  OUTs= b2s + F_DIM;                  // [128] accumulator
  int*    NL  = (int*)(OUTs + F_DIM);         // [64]  neighbor row offsets

  const int tid  = threadIdx.x;
  const int lane = tid & 31;
  const int wave = tid >> 5;
  const int ba0  = blockIdx.x * ITER;         // first (b,a) pair of this block
  const int b    = ba0 >> 9;                  // frame; invariant (512 % ITER == 0)
  const float* feat_b = features + (size_t)b * A_ATOMS * F_DIM;

  // ---- Weight + bias staging (once per block)
  if (PREP) {
#if HAVE_ASYNC_LDS
    AS1 v4i* g = (AS1 v4i*)((const AS1 char*)wsw + tid * 16);
    AS3 v4i* l = (AS3 v4i*)((AS3 char*)W1t + tid * 16);
#pragma unroll
    for (int j = 0; j < (int)(WS_BYTES / 16 / BDIM); ++j)   // 12 async b128
      __builtin_amdgcn_global_load_async_to_lds_b128(g + j * BDIM, l + j * BDIM, 0, 0);
#else
    const uint4* src = (const uint4*)wsw;
    uint4*       dst = (uint4*)W1t;           // W1t|W2t contiguous
#pragma unroll
    for (int i = tid; i < (int)(WS_BYTES / 16); i += BDIM) dst[i] = src[i];
#endif
  } else {
    for (int i = tid; i < W1T_ELEMS; i += BDIM) {
      const int gg = i >> 7, f = i & 127;
      W1t[f * G_DIM + gg] = (__bf16)W1[i];
    }
    for (int i = tid; i < W2T_ELEMS; i += BDIM) {
      const int c = i >> 7, f = i & 127;
      W2t[f * F_DIM + c] = (__bf16)W2[i];
    }
  }
  if (tid < F_DIM) { b1s[tid] = b1[tid]; b2s[tid] = b2[tid]; }

  // Wave tiling: wave owns row-tile mt (16 neighbor rows) x 4 column tiles.
  const int mt   = wave >> 1;
  const int row0 = mt * 16;
  const int ct0  = (wave & 1) * 4;
  const int half = lane >> 4;
  const int ncol = lane & 15;

#pragma unroll 1
  for (int it = 0; it < ITER; ++it) {
    const int ba = ba0 + it;
    const float*     rbf_t = rbf   + (size_t)ba * NNBR * G_DIM;
    const long long* nl_t  = nlist + (size_t)ba * NNBR;

    // ---- Stage: rbf tile (16 floats/thread -> 16 bf16, two b128 DS writes)
    {
      const float4* rv = (const float4*)(rbf_t) + tid * 4;
      union { __bf16 h[16]; uint4 q[2]; } t;
      float4 x0 = rv[0], x1 = rv[1], x2 = rv[2], x3 = rv[3];
      t.h[0]=(__bf16)x0.x; t.h[1]=(__bf16)x0.y; t.h[2]=(__bf16)x0.z; t.h[3]=(__bf16)x0.w;
      t.h[4]=(__bf16)x1.x; t.h[5]=(__bf16)x1.y; t.h[6]=(__bf16)x1.z; t.h[7]=(__bf16)x1.w;
      t.h[8]=(__bf16)x2.x; t.h[9]=(__bf16)x2.y; t.h[10]=(__bf16)x2.z; t.h[11]=(__bf16)x2.w;
      t.h[12]=(__bf16)x3.x; t.h[13]=(__bf16)x3.y; t.h[14]=(__bf16)x3.z; t.h[15]=(__bf16)x3.w;
      ((uint4*)Ab)[tid * 2 + 0] = t.q[0];
      ((uint4*)Ab)[tid * 2 + 1] = t.q[1];
    }
    if (tid < NNBR)  NL[tid]  = (int)nl_t[tid] * F_DIM;
    if (tid < F_DIM) OUTs[tid] = 0.f;
    // Prefetch next iteration's rbf tile into cache (global_prefetch_b8).
    if (it + 1 < ITER)
      __builtin_prefetch(rbf_t + NNBR * G_DIM + tid * 16, 0, 0);
#if HAVE_ASYNC_LDS
    if (PREP && it == 0) __builtin_amdgcn_s_wait_asynccnt(0);
#endif
    __syncthreads();

    // ---- GEMM1: H1 = tanh(rbf_bf16 @ W1_bf16 + b1), K = 64.
    {
      const v16bf a0 = load_a_frag(Ab, row0, 0,  G_DIM, lane);
      const v16bf a1 = load_a_frag(Ab, row0, 32, G_DIM, lane);
#pragma unroll
      for (int t = 0; t < 4; ++t) {
        const int n0 = (ct0 + t) * 16;
        v8f c = {};
        c = __builtin_amdgcn_wmma_f32_16x16x32_bf16(false, a0, false,
                load_b_frag(W1t, n0, 0,  G_DIM, lane), (short)0, c, false, false);
        c = __builtin_amdgcn_wmma_f32_16x16x32_bf16(false, a1, false,
                load_b_frag(W1t, n0, 32, G_DIM, lane), (short)0, c, false, false);
        const int   f    = n0 + ncol;
        const float bias = b1s[f];
#pragma unroll
        for (int r = 0; r < 8; ++r) {
          const int row = row0 + r + half * 8;   // C layout: M = r (+8 hi half)
          H1[row * F_DIM + f] = (__bf16)TANHF(c[r] + bias);
        }
      }
    }
    __syncthreads();

    // ---- GEMM2: CF = H1 @ W2 + b2, K = 128, fused gather-multiply-reduce.
    {
      v16bf h[4];
#pragma unroll
      for (int kk = 0; kk < 4; ++kk)
        h[kk] = load_a_frag(H1, row0, kk * 32, F_DIM, lane);

      int nloff[8];                              // neighbor row offsets, reused x4
#pragma unroll
      for (int r = 0; r < 8; ++r) nloff[r] = NL[row0 + r + half * 8];

#pragma unroll
      for (int t = 0; t < 4; ++t) {
        const int n0 = (ct0 + t) * 16;
        v8f c = {};
#pragma unroll
        for (int kk = 0; kk < 4; ++kk)
          c = __builtin_amdgcn_wmma_f32_16x16x32_bf16(false, h[kk], false,
                  load_b_frag(W2t, n0, kk * 32, F_DIM, lane), (short)0, c, false, false);
        const int   f    = n0 + ncol;
        const float bias = b2s[f];
        float partial = 0.f;
#pragma unroll
        for (int r = 0; r < 8; ++r) {
          const float cf = c[r] + bias;
          const float fv = feat_b[nloff[r] + f]; // gathered feature (coalesced 64B)
          partial = fmaf(cf, fv, partial);
        }
        partial += __shfl_xor(partial, 16, 32);  // fold lane-halves (M 0-7 + 8-15)
        if (half == 0) atomicAdd(&OUTs[f], partial);   // ds_add_f32
      }
    }
    __syncthreads();

    if (tid < F_DIM) out[(size_t)ba * F_DIM + tid] = OUTs[tid];
    // No extra barrier needed: only tid<128 touch OUTs (store then re-zero in
    // program order); Ab/NL rewrites are fenced by the post-stage barrier.
  }
}

template __global__ void cfconv_wmma_bf16<true>(const float*, const float*,
    const long long*, const float*, const float*, const float*, const float*,
    const __bf16*, float*);
template __global__ void cfconv_wmma_bf16<false>(const float*, const float*,
    const long long*, const float*, const float*, const float*, const float*,
    const __bf16*, float*);

extern "C" void kernel_launch(void* const* d_in, const int* in_sizes, int n_in,
                              void* d_out, int out_size, void* d_ws, size_t ws_size,
                              hipStream_t stream) {
  const float*     features = (const float*)d_in[0];     // [B,A,F]   f32
  const float*     rbf      = (const float*)d_in[1];     // [B,A,N,G] f32
  const long long* nlist    = (const long long*)d_in[2]; // [B,A,N]   i64
  const float*     W1       = (const float*)d_in[3];     // [G,F]
  const float*     b1       = (const float*)d_in[4];     // [F]
  const float*     W2       = (const float*)d_in[5];     // [F,F]
  const float*     b2       = (const float*)d_in[6];     // [F]
  float*           out      = (float*)d_out;             // [B,A,F]

  const size_t smem =
      (size_t)(W1T_ELEMS + W2T_ELEMS + NNBR * G_DIM + NNBR * F_DIM) * sizeof(__bf16)
      + (size_t)(3 * F_DIM) * sizeof(float)
      + (size_t)NNBR * sizeof(int);   // ~75.5 KB; >=4 blocks per 320 KB WGP

  const dim3 grid(B_FRM * A_ATOMS / ITER);
  if (ws_size >= (size_t)WS_BYTES) {
    __bf16* ws = (__bf16*)d_ws;
    hipLaunchKernelGGL(cfconv_prep_weights, dim3(8), dim3(BDIM), 0, stream,
                       W1, W2, ws);
    hipLaunchKernelGGL((cfconv_wmma_bf16<true>), grid, dim3(BDIM), smem, stream,
                       features, rbf, nlist, W1, b1, W2, b2, ws, out);
  } else {
    hipLaunchKernelGGL((cfconv_wmma_bf16<false>), grid, dim3(BDIM), smem, stream,
                       features, rbf, nlist, W1, b1, W2, b2, (const __bf16*)nullptr, out);
  }
}
